// MixtureOfExperts_76304388981194
// MI455X (gfx1250) — matrix operand order repaired and checked by reading, not compiled, not run on hardware
//
#include <hip/hip_runtime.h>
#include <math.h>

#define B_    4
#define T_    2048
#define D_    1024
#define F_    2048
#define E_    8
#define NTOK  (B_ * T_)          // 8192 tokens
#define NROWS (NTOK * 2)         // total expert-assigned rows (K=2, exact)

typedef __attribute__((ext_vector_type(16))) __bf16 v16bf;
typedef __attribute__((ext_vector_type(8)))  float  v8f;
typedef unsigned short u16;
typedef unsigned int   u32;

union Frag { uint4 q[2]; u32 u[8]; v16bf v; };
static_assert(sizeof(Frag) == 32, "frag size");

// ---------------- workspace layout (bytes, all 64-aligned) ----------------
#define WS_COUNTS   0ull                                  // E ints
#define WS_OFFSETS  64ull                                 // E+1 ints
#define WS_LIST_TOK 256ull                                // E*NTOK ints
#define WS_LIST_W   (WS_LIST_TOK + (size_t)E_ * NTOK * 4) // E*NTOK floats
#define WS_XS_HI    (WS_LIST_W   + (size_t)E_ * NTOK * 4) // NTOK*D bf16
#define WS_XS_LO    (WS_XS_HI  + (size_t)NTOK * D_ * 2)
#define WS_W1T_HI   (WS_XS_LO  + (size_t)NTOK * D_ * 2)   // E*F*D bf16 (transposed)
#define WS_W1T_LO   (WS_W1T_HI + (size_t)E_ * D_ * F_ * 2)
#define WS_W2T_HI   (WS_W1T_LO + (size_t)E_ * D_ * F_ * 2) // E*D*F bf16 (transposed)
#define WS_W2T_LO   (WS_W2T_HI + (size_t)E_ * D_ * F_ * 2)
#define WS_H_HI     (WS_W2T_LO + (size_t)E_ * D_ * F_ * 2) // NROWS*F bf16
#define WS_H_LO     (WS_H_HI   + (size_t)NROWS * F_ * 2)

// ---------------- bf16 split helpers (RNE, finite inputs) ----------------
__device__ inline u16 f2bf(float f) {
  u32 u = __builtin_bit_cast(u32, f);
  u = (u + 0x7FFFu + ((u >> 16) & 1u)) >> 16;
  return (u16)u;
}
__device__ inline float bf2f(u16 h) {
  u32 u = ((u32)h) << 16;
  return __builtin_bit_cast(float, u);
}
__device__ inline void split2(float f, u16& hi, u16& lo) {
  u16 h = f2bf(f);
  hi = h;
  lo = f2bf(f - bf2f(h));
}

// ---------------- router: logits -> top2 -> softmax -> expert lists ----------------
__global__ __launch_bounds__(256) void moe_router(
    const float* __restrict__ x, const float* __restrict__ gw,
    const float* __restrict__ gb, int* __restrict__ counts,
    int* __restrict__ list_tok, float* __restrict__ list_w) {
  const int t    = (blockIdx.x * 256 + threadIdx.x) >> 5;
  const int lane = threadIdx.x & 31;
  if (t >= NTOK) return;
  const float* xr = x + (size_t)t * D_;

  float acc[E_];
#pragma unroll
  for (int e = 0; e < E_; ++e) acc[e] = 0.f;
  for (int d = lane; d < D_; d += 32) {
    const float xv = xr[d];
    const float* gr = gw + (size_t)d * E_;
#pragma unroll
    for (int e = 0; e < E_; ++e) acc[e] += xv * gr[e];
  }
#pragma unroll
  for (int off = 16; off >= 1; off >>= 1) {
#pragma unroll
    for (int e = 0; e < E_; ++e) acc[e] += __shfl_xor(acc[e], off, 32);
  }
  if (lane == 0) {
    float v[E_];
#pragma unroll
    for (int e = 0; e < E_; ++e) v[e] = acc[e] + gb[e];
    int i0 = 0;
#pragma unroll
    for (int e = 1; e < E_; ++e) if (v[e] > v[i0]) i0 = e;
    int i1 = (i0 == 0) ? 1 : 0;
#pragma unroll
    for (int e = 0; e < E_; ++e) if (e != i1 && e != i0 && v[e] > v[i1]) i1 = e;
    const float w0 = 1.f / (1.f + expf(v[i1] - v[i0]));
    const float w1 = 1.f - w0;
    int p0 = atomicAdd(&counts[i0], 1);
    list_tok[i0 * NTOK + p0] = t;  list_w[i0 * NTOK + p0] = w0;
    int p1 = atomicAdd(&counts[i1], 1);
    list_tok[i1 * NTOK + p1] = t;  list_w[i1 * NTOK + p1] = w1;
  }
}

__global__ void moe_scan(const int* __restrict__ counts, int* __restrict__ offsets) {
  if (blockIdx.x == 0 && threadIdx.x == 0) {
    int s = 0;
    for (int e = 0; e < E_; ++e) { offsets[e] = s; s += counts[e]; }
    offsets[E_] = s;
  }
}

// ---------------- pre-pass: split x into bf16 hi/lo planes ----------------
__global__ __launch_bounds__(256) void split_x(
    const float* __restrict__ src, u16* __restrict__ dhi, u16* __restrict__ dlo,
    int n4) {
  const int i = blockIdx.x * 256 + threadIdx.x;
  if (i >= n4) return;
  const float4 f = ((const float4*)src)[i];
  u16 h0, l0, h1, l1, h2, l2, h3, l3;
  split2(f.x, h0, l0); split2(f.y, h1, l1);
  split2(f.z, h2, l2); split2(f.w, h3, l3);
  ((uint2*)dhi)[i] = make_uint2((u32)h0 | ((u32)h1 << 16), (u32)h2 | ((u32)h3 << 16));
  ((uint2*)dlo)[i] = make_uint2((u32)l0 | ((u32)l1 << 16), (u32)l2 | ((u32)l3 << 16));
}

// ---------------- pre-pass: per-expert transpose + split ----------------
// src: [E][R][C] fp32 -> dhi/dlo: [E][C][R] bf16
__global__ __launch_bounds__(256) void transpose_split(
    const float* __restrict__ src, u16* __restrict__ dhi, u16* __restrict__ dlo,
    int R, int C) {
  __shared__ u16 th[32][33];
  __shared__ u16 tl[32][33];
  const int e  = blockIdx.z;
  const int c0 = blockIdx.x * 32, r0 = blockIdx.y * 32;
  const size_t base = (size_t)e * R * C;
  const int t  = threadIdx.x;
  const int cl = t & 31, rl = t >> 5;
#pragma unroll
  for (int rr = 0; rr < 4; ++rr) {
    const int r = rl + rr * 8;
    const float v = src[base + (size_t)(r0 + r) * C + (c0 + cl)];
    u16 hi, lo; split2(v, hi, lo);
    th[r][cl] = hi; tl[r][cl] = lo;
  }
  __syncthreads();
  const int cw = t >> 4, rp = (t & 15) * 2;
#pragma unroll
  for (int cc = 0; cc < 2; ++cc) {
    const int c = cw + cc * 16;
    const u32 uh = (u32)th[rp][c] | ((u32)th[rp + 1][c] << 16);
    const u32 ul = (u32)tl[rp][c] | ((u32)tl[rp + 1][c] << 16);
    const size_t o = base + (size_t)(c0 + c) * R + (r0 + rp);
    *(u32*)&dhi[o] = uh;
    *(u32*)&dlo[o] = ul;
  }
}

// ---------------- tile machinery ----------------
// Block tile 128(M) x 128(N) x 32(K); 8 waves 4(M)x2(N); per wave 2x4 16x16 accs.
// LDS pitch 40 ushorts (80B rows): 16B-aligned rows, conflict-free 20-bank stride.
// bf16x3: acc += Ahi*Bhi + Ahi*Blo + Alo*Bhi (fp32 accumulate).
#define LDS_PITCH 40

__device__ inline void load_a_frag(const u16 (*S)[LDS_PITCH], int lr, int half, Frag& f) {
  const u16* rp = &S[lr][0];
  f.q[0] = *(const uint4*)(rp + half * 8);        // K = half*8 .. +7
  f.q[1] = *(const uint4*)(rp + 16 + half * 8);   // K = 16+half*8 .. +7
}
__device__ inline void load_b_frag(const u16 (*S)[LDS_PITCH], int nr, int half, Frag& f) {
  const u16* rp = &S[nr][0];
  f.q[0] = *(const uint4*)(rp + half * 8);        // K = half*8 .. +7
  f.q[1] = *(const uint4*)(rp + 16 + half * 8);   // K = 16+half*8 .. +7
}

#define WMMA3(ACC, AH, AL, BH, BL)                                             \
  ACC = __builtin_amdgcn_wmma_f32_16x16x32_bf16(false, (AH).v, false, (BH).v,  \
                                                (short)0, ACC, false, false);  \
  ACC = __builtin_amdgcn_wmma_f32_16x16x32_bf16(false, (AH).v, false, (BL).v,  \
                                                (short)0, ACC, false, false);  \
  ACC = __builtin_amdgcn_wmma_f32_16x16x32_bf16(false, (AL).v, false, (BH).v,  \
                                                (short)0, ACC, false, false);

// ---------------- FFN1: h = gelu(x_gathered @ w1[e] + b1[e]) ----------------
__global__ __launch_bounds__(256) void moe_ffn1(
    const u16* __restrict__ xs_hi, const u16* __restrict__ xs_lo,
    const u16* __restrict__ w1t_hi, const u16* __restrict__ w1t_lo,
    const float* __restrict__ b1, const int* __restrict__ counts,
    const int* __restrict__ offsets, const int* __restrict__ list_tok,
    u16* __restrict__ h_hi, u16* __restrict__ h_lo) {
  const int e  = blockIdx.z;
  const int Me = counts[e];
  const int m0 = blockIdx.x * 128;
  if (m0 >= Me) return;
  const int n0  = blockIdx.y * 128;
  const int ofs = offsets[e];
  const int* toks = list_tok + e * NTOK;

  __shared__ u16 As_hi[128][LDS_PITCH];
  __shared__ u16 As_lo[128][LDS_PITCH];
  __shared__ u16 Bs_hi[128][LDS_PITCH];
  __shared__ u16 Bs_lo[128][LDS_PITCH];

  const int tid  = threadIdx.x;
  const int lane = tid & 31, wave = tid >> 5;
  const int wm = wave >> 1, wn = wave & 1;
  const int half = lane >> 4, r = lane & 15;

  v8f acc[2][4];
#pragma unroll
  for (int a = 0; a < 2; ++a)
#pragma unroll
    for (int b = 0; b < 4; ++b) acc[a][b] = (v8f){0, 0, 0, 0, 0, 0, 0, 0};

  const int sr = tid >> 1, sh = tid & 1;   // staging: row, 16-elem half
  const int tok = (m0 + sr < Me) ? toks[m0 + sr] : -1;
  const size_t aoff = (tok >= 0) ? ((size_t)tok * D_ + sh * 16) : 0;
  const size_t boff = ((size_t)e * F_ + n0 + sr) * D_ + sh * 16;
  const uint4 z4 = make_uint4(0, 0, 0, 0);

  for (int k0 = 0; k0 < D_; k0 += 32) {
    if (k0 + 32 < D_) {   // prefetch next k-slice (global_prefetch_b8)
      if (tok >= 0) __builtin_prefetch(xs_hi + aoff + k0 + 32, 0, 0);
      __builtin_prefetch(w1t_hi + boff + k0 + 32, 0, 0);
    }
    __syncthreads();
    {   // stage A (pure bf16 copies)
      uint4 ah0 = z4, ah1 = z4, al0 = z4, al1 = z4;
      if (tok >= 0) {
        const uint4* ph = (const uint4*)(xs_hi + aoff + k0);
        const uint4* pl = (const uint4*)(xs_lo + aoff + k0);
        ah0 = ph[0]; ah1 = ph[1]; al0 = pl[0]; al1 = pl[1];
      }
      *(uint4*)&As_hi[sr][sh * 16]     = ah0;
      *(uint4*)&As_hi[sr][sh * 16 + 8] = ah1;
      *(uint4*)&As_lo[sr][sh * 16]     = al0;
      *(uint4*)&As_lo[sr][sh * 16 + 8] = al1;
    }
    {   // stage B (already transposed in workspace: [n][k] contiguous)
      const uint4* ph = (const uint4*)(w1t_hi + boff + k0);
      const uint4* pl = (const uint4*)(w1t_lo + boff + k0);
      *(uint4*)&Bs_hi[sr][sh * 16]     = ph[0];
      *(uint4*)&Bs_hi[sr][sh * 16 + 8] = ph[1];
      *(uint4*)&Bs_lo[sr][sh * 16]     = pl[0];
      *(uint4*)&Bs_lo[sr][sh * 16 + 8] = pl[1];
    }
    __syncthreads();

    Frag ah2[2], al2[2], bh2[4], bl2[4];
#pragma unroll
    for (int ms = 0; ms < 2; ++ms) {
      const int lr = wm * 32 + ms * 16 + r;
      load_a_frag(As_hi, lr, half, ah2[ms]);
      load_a_frag(As_lo, lr, half, al2[ms]);
    }
#pragma unroll
    for (int ns = 0; ns < 4; ++ns) {
      const int nr = wn * 64 + ns * 16 + r;
      load_b_frag(Bs_hi, nr, half, bh2[ns]);
      load_b_frag(Bs_lo, nr, half, bl2[ns]);
    }
#pragma unroll
    for (int ms = 0; ms < 2; ++ms)
#pragma unroll
      for (int ns = 0; ns < 4; ++ns) {
        WMMA3(acc[ms][ns], ah2[ms], al2[ms], bh2[ns], bl2[ns]);
      }
  }

  // epilogue: + b1, exact-erf GELU, store split-bf16 h
#pragma unroll
  for (int ms = 0; ms < 2; ++ms)
#pragma unroll
    for (int ns = 0; ns < 4; ++ns) {
      const int col  = n0 + wn * 64 + ns * 16 + r;
      const float bv = b1[e * F_ + col];
#pragma unroll
      for (int vv = 0; vv < 8; ++vv) {
        const int rowl = m0 + wm * 32 + ms * 16 + 8 * half + vv;
        if (rowl < Me) {
          const float tv = acc[ms][ns][vv] + bv;
          const float g  = 0.5f * tv * (1.f + erff(tv * 0.70710678118654752f));
          u16 hi, lo; split2(g, hi, lo);
          const size_t o = (size_t)(ofs + rowl) * F_ + col;
          h_hi[o] = hi; h_lo[o] = lo;
        }
      }
    }
}

// ---------------- FFN2: out[tok] += w * (h @ w2[e] + b2[e]) ----------------
__global__ __launch_bounds__(256) void moe_ffn2(
    const u16* __restrict__ h_hi, const u16* __restrict__ h_lo,
    const u16* __restrict__ w2t_hi, const u16* __restrict__ w2t_lo,
    const float* __restrict__ b2, const int* __restrict__ counts,
    const int* __restrict__ offsets, const int* __restrict__ list_tok,
    const float* __restrict__ list_w, float* __restrict__ out) {
  const int e  = blockIdx.z;
  const int Me = counts[e];
  const int m0 = blockIdx.x * 128;
  if (m0 >= Me) return;
  const int n0  = blockIdx.y * 128;
  const int ofs = offsets[e];
  const int*   toks = list_tok + e * NTOK;
  const float* wts  = list_w   + e * NTOK;

  __shared__ u16 As_hi[128][LDS_PITCH];
  __shared__ u16 As_lo[128][LDS_PITCH];
  __shared__ u16 Bs_hi[128][LDS_PITCH];
  __shared__ u16 Bs_lo[128][LDS_PITCH];

  const int tid  = threadIdx.x;
  const int lane = tid & 31, wave = tid >> 5;
  const int wm = wave >> 1, wn = wave & 1;
  const int half = lane >> 4, r = lane & 15;

  v8f acc[2][4];
#pragma unroll
  for (int a = 0; a < 2; ++a)
#pragma unroll
    for (int b = 0; b < 4; ++b) acc[a][b] = (v8f){0, 0, 0, 0, 0, 0, 0, 0};

  const int sr = tid >> 1, sh = tid & 1;
  const bool av = (m0 + sr) < Me;
  const size_t aoff = (size_t)(ofs + m0 + sr) * F_ + sh * 16;
  const size_t boff = ((size_t)e * D_ + n0 + sr) * F_ + sh * 16;
  const uint4 z4 = make_uint4(0, 0, 0, 0);

  for (int k0 = 0; k0 < F_; k0 += 32) {
    if (k0 + 32 < F_) {
      if (av) __builtin_prefetch(h_hi + aoff + k0 + 32, 0, 0);
      __builtin_prefetch(w2t_hi + boff + k0 + 32, 0, 0);
    }
    __syncthreads();
    {
      uint4 ah0 = z4, ah1 = z4, al0 = z4, al1 = z4;
      if (av) {
        const uint4* ph = (const uint4*)(h_hi + aoff + k0);
        const uint4* pl = (const uint4*)(h_lo + aoff + k0);
        ah0 = ph[0]; ah1 = ph[1]; al0 = pl[0]; al1 = pl[1];
      }
      *(uint4*)&As_hi[sr][sh * 16]     = ah0;
      *(uint4*)&As_hi[sr][sh * 16 + 8] = ah1;
      *(uint4*)&As_lo[sr][sh * 16]     = al0;
      *(uint4*)&As_lo[sr][sh * 16 + 8] = al1;
    }
    {
      const uint4* ph = (const uint4*)(w2t_hi + boff + k0);
      const uint4* pl = (const uint4*)(w2t_lo + boff + k0);
      *(uint4*)&Bs_hi[sr][sh * 16]     = ph[0];
      *(uint4*)&Bs_hi[sr][sh * 16 + 8] = ph[1];
      *(uint4*)&Bs_lo[sr][sh * 16]     = pl[0];
      *(uint4*)&Bs_lo[sr][sh * 16 + 8] = pl[1];
    }
    __syncthreads();

    Frag ah2[2], al2[2], bh2[4], bl2[4];
#pragma unroll
    for (int ms = 0; ms < 2; ++ms) {
      const int lr = wm * 32 + ms * 16 + r;
      load_a_frag(As_hi, lr, half, ah2[ms]);
      load_a_frag(As_lo, lr, half, al2[ms]);
    }
#pragma unroll
    for (int ns = 0; ns < 4; ++ns) {
      const int nr = wn * 64 + ns * 16 + r;
      load_b_frag(Bs_hi, nr, half, bh2[ns]);
      load_b_frag(Bs_lo, nr, half, bl2[ns]);
    }
#pragma unroll
    for (int ms = 0; ms < 2; ++ms)
#pragma unroll
      for (int ns = 0; ns < 4; ++ns) {
        WMMA3(acc[ms][ns], ah2[ms], al2[ms], bh2[ns], bl2[ns]);
      }
  }

  // epilogue: + b2, scale by routing weight, atomic combine into out
#pragma unroll
  for (int ms = 0; ms < 2; ++ms)
#pragma unroll
    for (int ns = 0; ns < 4; ++ns) {
      const int col  = n0 + wn * 64 + ns * 16 + r;
      const float bv = b2[e * D_ + col];
#pragma unroll
      for (int vv = 0; vv < 8; ++vv) {
        const int rowl = m0 + wm * 32 + ms * 16 + 8 * half + vv;
        if (rowl < Me) {
          const int   tk = toks[rowl];
          const float wt = wts[rowl];
          atomicAdd(&out[(size_t)tk * D_ + col], wt * (acc[ms][ns][vv] + bv));
        }
      }
    }
}

// ---------------- launch ----------------
extern "C" void kernel_launch(void* const* d_in, const int* in_sizes, int n_in,
                              void* d_out, int out_size, void* d_ws, size_t ws_size,
                              hipStream_t stream) {
  const float* x      = (const float*)d_in[0];
  const float* gate_w = (const float*)d_in[1];
  const float* gate_b = (const float*)d_in[2];
  const float* w1     = (const float*)d_in[3];
  const float* b1     = (const float*)d_in[4];
  const float* w2     = (const float*)d_in[5];
  const float* b2     = (const float*)d_in[6];
  float* out = (float*)d_out;

  char* ws = (char*)d_ws;
  int*   counts   = (int*)(ws + WS_COUNTS);
  int*   offsets  = (int*)(ws + WS_OFFSETS);
  int*   list_tok = (int*)(ws + WS_LIST_TOK);
  float* list_w   = (float*)(ws + WS_LIST_W);
  u16*   xs_hi    = (u16*)(ws + WS_XS_HI);
  u16*   xs_lo    = (u16*)(ws + WS_XS_LO);
  u16*   w1t_hi   = (u16*)(ws + WS_W1T_HI);
  u16*   w1t_lo   = (u16*)(ws + WS_W1T_LO);
  u16*   w2t_hi   = (u16*)(ws + WS_W2T_HI);
  u16*   w2t_lo   = (u16*)(ws + WS_W2T_LO);
  u16*   h_hi     = (u16*)(ws + WS_H_HI);
  u16*   h_lo     = (u16*)(ws + WS_H_LO);

  hipMemsetAsync(counts, 0, 64, stream);
  hipMemsetAsync(out, 0, (size_t)out_size * sizeof(float), stream);

  moe_router<<<NTOK / 8, 256, 0, stream>>>(x, gate_w, gate_b, counts, list_tok, list_w);
  moe_scan<<<1, 32, 0, stream>>>(counts, offsets);

  // pre-pass: split/transpose into bf16 hi/lo planes
  const int n4 = NTOK * D_ / 4;
  split_x<<<(n4 + 255) / 256, 256, 0, stream>>>(x, xs_hi, xs_lo, n4);
  dim3 gt1(F_ / 32, D_ / 32, E_);   // w1 [E][D][F] -> w1t [E][F][D]
  transpose_split<<<gt1, 256, 0, stream>>>(w1, w1t_hi, w1t_lo, D_, F_);
  dim3 gt2(D_ / 32, F_ / 32, E_);   // w2 [E][F][D] -> w2t [E][D][F]
  transpose_split<<<gt2, 256, 0, stream>>>(w2, w2t_hi, w2t_lo, F_, D_);

  dim3 g1(NTOK / 128, F_ / 128, E_);   // worst-case Me; blocks early-exit on count
  moe_ffn1<<<g1, 256, 0, stream>>>(xs_hi, xs_lo, w1t_hi, w1t_lo, b1,
                                   counts, offsets, list_tok, h_hi, h_lo);
  dim3 g2(NTOK / 128, D_ / 128, E_);
  moe_ffn2<<<g2, 256, 0, stream>>>(h_hi, h_lo, w2t_hi, w2t_lo, b2,
                                   counts, offsets, list_tok, list_w, out);
}
